// PAM_40278203302431
// MI455X (gfx1250) — matrix operand order repaired and checked by reading, not compiled, not run on hardware
//
#include <hip/hip_runtime.h>

// MI455X / gfx1250, wave32. One workgroup (8 waves, 256 threads) per (b,g,h)
// slice. Memory-bound problem (~100 MB HBM traffic, ~1 GFLOP) -> fuse
// correlation GEMM + shifted cost volume + softmax flag in one pass, keep the
// 128x128 f matrix resident in LDS, use v_wmma_f32_16x16x32_f16 (K=32 == Nc,
// one WMMA per 16x16 tile).

typedef _Float16 v16h __attribute__((ext_vector_type(16)));
typedef _Float16 v8h  __attribute__((ext_vector_type(8)));
typedef float    v8f  __attribute__((ext_vector_type(8)));

#define BB 2
#define CC 256
#define HH 64
#define WW 128
#define GG 8
#define NC 32          // channels per group = WMMA K
#define LDK 40         // padded f16 row stride (80 B, 16 B aligned)

__global__ __launch_bounds__(256)
void pam_corr_volume_kernel(const float* __restrict__ xl,
                            const float* __restrict__ xr,
                            float* __restrict__ out)
{
    extern __shared__ unsigned char smem[];
    // Phase 1 layout: A (xr, [w][c] f16) then B (xl, [v][c] f16)
    _Float16* ldsA = (_Float16*)smem;                       // 128*40*2 = 10240 B
    _Float16* ldsB = (_Float16*)(smem + WW * LDK * 2);      // 10240 B
    // Phase 2 layout (overlaid after all frag reads complete):
    float* fbuf = (float*)smem;                             // [128][128] = 65536 B

    const int t    = threadIdx.x;
    const int lane = t & 31;
    const int wave = t >> 5;
    const int mrow = lane & 15;      // row/col within half-wave
    const int hi   = lane >> 4;      // 0: lanes 0-15, 1: lanes 16-31

    const int h  = blockIdx.x % HH;
    const int bg = blockIdx.x / HH;  // b*G+g, 0..15

    // ---------------- Stage slices to LDS, transposing [c][pos] -> [pos][c],
    // converting f32 -> f16. Coalesced global reads (consecutive t -> w).
    const long sliceBase = (long)bg * NC * (HH * WW) + (long)h * WW;
    #pragma unroll
    for (int i = 0; i < (NC * WW) / 256; ++i) {
        int idx = t + i * 256;
        int c = idx >> 7;            // / WW
        int w = idx & (WW - 1);
        float vr = xr[sliceBase + (long)c * (HH * WW) + w];
        float vl = xl[sliceBase + (long)c * (HH * WW) + w];
        ldsA[w * LDK + c] = (_Float16)vr;   // A: rows = query pos w
        ldsB[w * LDK + c] = (_Float16)vl;   // B: cols = key pos v
    }
    __syncthreads();

    // ---------------- WMMA phase: wave computes rows m0..m0+15 x all 8 col tiles.
    const int m0 = wave * 16;

    // A fragment (16x32 f16): lane<16 holds K=0..7,16..23; lane>=16 K=8..15,24..31
    v16h afrag;
    {
        const _Float16* ap = ldsA + (m0 + mrow) * LDK + (hi ? 8 : 0);
        v8h lo = *(const v8h*)(ap);
        v8h hq = *(const v8h*)(ap + 16);
        #pragma unroll
        for (int i = 0; i < 8; ++i) { afrag[i] = lo[i]; afrag[i + 8] = hq[i]; }
    }

    v8f acc[8];
    #pragma unroll
    for (int nt = 0; nt < 8; ++nt) {
        // B fragment (32x16 f16): lane<16 (col N) holds K=0..7,16..23; lane>=16 K=8..15,24..31
        v16h bfrag;
        const _Float16* bp = ldsB + (nt * 16 + mrow) * LDK + (hi ? 8 : 0);
        v8h lo = *(const v8h*)(bp);
        v8h hq = *(const v8h*)(bp + 16);
        #pragma unroll
        for (int i = 0; i < 8; ++i) { bfrag[i] = lo[i]; bfrag[i + 8] = hq[i]; }

        v8f c = {};
        acc[nt] = __builtin_amdgcn_wmma_f32_16x16x32_f16(
            /*neg_a=*/false, afrag, /*neg_b=*/false, bfrag,
            /*c_mod=*/(short)0, c, /*reuse_a=*/false, /*reuse_b=*/false);
    }
    __syncthreads();   // all ldsA/ldsB reads done -> safe to overlay fbuf

    // Store f tiles (scaled by 1/Nc). C/D layout: VGPR r, lane<16 -> M=r,
    // lane>=16 -> M=8+r; N = lane&15.
    #pragma unroll
    for (int nt = 0; nt < 8; ++nt) {
        #pragma unroll
        for (int r = 0; r < 8; ++r) {
            int row = m0 + r + (hi ? 8 : 0);
            int col = nt * 16 + mrow;
            fbuf[row * WW + col] = acc[nt][r] * (1.0f / NC);
        }
    }
    __syncthreads();

    // ---------------- Shifted cost volume: volume[bg, j, h, k] = f[j][j+k] (j+k<W)
    // out index: ((bg*W + j)*H + h)*W + k ; contiguous in k -> coalesced stores.
    const long volBase = (long)bg * WW * HH * WW + (long)h * WW;
    #pragma unroll
    for (int i = 0; i < (WW * WW) / 256; ++i) {
        int idx = t + i * 256;
        int j = idx >> 7;
        int k = idx & (WW - 1);
        float v = (j + k < WW) ? fbuf[j * WW + (j + k)] : 0.0f;
        out[volBase + (long)j * (HH * WW) + k] = v;
    }

    // ---------------- V flag: sum of softmax(f[j,:]) > 1  (numerical artifact)
    if (t < WW) {
        const float* row = fbuf + t * WW;
        float mx = row[0];
        for (int v = 1; v < WW; ++v) mx = fmaxf(mx, row[v]);
        float s = 0.0f;
        for (int v = 0; v < WW; ++v) s += __expf(row[v] - mx);
        float psum = 0.0f;
        for (int v = 0; v < WW; ++v) psum += __expf(row[v] - mx) / s;
        const long vBase = (long)BB * GG * WW * HH * WW;   // after volume
        out[vBase + (long)bg * (HH * WW) + (long)h * WW + t] =
            (psum > 1.0f) ? 1.0f : 0.0f;
    }
}

extern "C" void kernel_launch(void* const* d_in, const int* in_sizes, int n_in,
                              void* d_out, int out_size, void* d_ws, size_t ws_size,
                              hipStream_t stream) {
    const float* x_l = (const float*)d_in[0];
    const float* x_r = (const float*)d_in[1];
    // d_in[2] = num_groups (compile-time constant GG here)
    float* out = (float*)d_out;

    dim3 grid(BB * GG * HH);   // 1024 slices
    dim3 block(256);           // 8 wave32s
    size_t smemBytes = (size_t)WW * WW * sizeof(float);  // 64 KB (covers both phases)
    hipLaunchKernelGGL(pam_corr_volume_kernel, grid, block, smemBytes, stream,
                       x_l, x_r, out);
}
